// TriplaneDecoder_25056839205926
// MI455X (gfx1250) — matrix operand order repaired and checked by reading, not compiled, not run on hardware
//
#include <hip/hip_runtime.h>
#include <hip/hip_bf16.h>

typedef _Float16 half16 __attribute__((ext_vector_type(16)));
typedef float    f32x8  __attribute__((ext_vector_type(8)));

#define BB   4
#define NN   100000
#define CC   64
#define HH   256
#define WW   256
#define HID  256
#define K_IN 192          // 3*C
#define NLAY 10           // input layer + 9 hidden

#define TM   64           // points per block
#define LDSK 272          // padded f16 row stride (544B = 34*16, 16B aligned rows)

#define W_IN_E   (HID * K_IN)      // 49152
#define W_HID_E  (9 * HID * HID)   // 589824
#define W_HEAD_E (16 * HID)        // 4096  (rows: 0-2 rgb, 3 den, 4-6 emb, 7-15 zero)
#define W_TOTAL  (W_IN_E + W_HID_E + W_HEAD_E)

// ---------------- prep: f32 weights -> f16 workspace ----------------
__global__ void prep_weights(const float* __restrict__ w_in,
                             const float* __restrict__ w_hid,
                             const float* __restrict__ w_rgb,
                             const float* __restrict__ w_den,
                             const float* __restrict__ w_emb,
                             _Float16* __restrict__ ws)
{
    int i = blockIdx.x * blockDim.x + threadIdx.x;
    if (i >= W_TOTAL) return;
    if (i < W_IN_E) { ws[i] = (_Float16)w_in[i]; return; }
    int j = i - W_IN_E;
    if (j < W_HID_E) { ws[i] = (_Float16)w_hid[j]; return; }
    int k = j - W_HID_E;
    int r = k / HID, c = k % HID;
    float v = 0.0f;
    if      (r < 3) v = w_rgb[r * HID + c];
    else if (r == 3) v = w_den[c];
    else if (r < 7) v = w_emb[(r - 4) * HID + c];
    ws[i] = (_Float16)v;
}

// ---------------- one dense layer: h <- relu(h @ W^T + b), WMMA ----------------
template <int K>
__device__ __forceinline__ void mlp_layer(const _Float16* __restrict__ Wt,
                                          const float*    __restrict__ bias,
                                          _Float16* hbuf,
                                          int waveid, int lo16, int hi, int mrow)
{
    constexpr int KF = K / 32;
    half16 afr[KF];
    const _Float16* arow = hbuf + mrow * LDSK;
    #pragma unroll
    for (int kf = 0; kf < KF; ++kf) {
        const int kb = kf * 32 + hi * 8;
        ((uint4*)&afr[kf])[0] = *(const uint4*)(arow + kb);
        ((uint4*)&afr[kf])[1] = *(const uint4*)(arow + kb + 16);
    }
    for (int nt = 0; nt < 16; ++nt) {
        const int n = nt * 16 + lo16;
        f32x8 acc = {0.f,0.f,0.f,0.f,0.f,0.f,0.f,0.f};
        const _Float16* wr = Wt + (size_t)n * K + hi * 8;
        #pragma unroll
        for (int kf = 0; kf < KF; ++kf) {
            half16 bfr;
            ((uint4*)&bfr)[0] = *(const uint4*)(wr + kf * 32);
            ((uint4*)&bfr)[1] = *(const uint4*)(wr + kf * 32 + 16);
            acc = __builtin_amdgcn_wmma_f32_16x16x32_f16(false, afr[kf], false, bfr,
                                                         (short)0, acc, false, false);
        }
        const float bv = bias[n];
        _Float16* ocol = hbuf + (waveid * 16 + hi * 8) * LDSK + n;
        #pragma unroll
        for (int v = 0; v < 8; ++v) {
            float r = acc[v] + bv;
            r = r > 0.f ? r : 0.f;
            ocol[v * LDSK] = (_Float16)r;
        }
    }
}

// ---------------- fused sample + MLP + heads ----------------
__global__ __launch_bounds__(128)
void triplane_mlp_kernel(const float* __restrict__ triplane,
                         const float* __restrict__ points,
                         const _Float16* __restrict__ w_in_h,
                         const float* __restrict__ b_in,
                         const _Float16* __restrict__ w_hid_h,
                         const float* __restrict__ b_hid,
                         const _Float16* __restrict__ w_head_h,
                         const float* __restrict__ b_rgb,
                         const float* __restrict__ b_den,
                         const float* __restrict__ b_emb,
                         float* __restrict__ out)
{
    __shared__ __attribute__((aligned(16))) _Float16 hbuf[TM * LDSK];

    const int tid = threadIdx.x;
    const int g0  = blockIdx.x * TM;

    // ---- phase 1: bilinear triplane sampling (border pad, align_corners) ----
    {
        const int pt = tid & 63;
        const int cg = tid >> 6;          // 0/1: channels [cg*32, cg*32+32)
        const int g  = g0 + pt;
        const int b  = g / NN;
        const float px = points[(size_t)g * 3 + 0];
        const float py = points[(size_t)g * 3 + 1];
        const float pz = points[(size_t)g * 3 + 2];
        #pragma unroll
        for (int pl = 0; pl < 3; ++pl) {
            const float gx = (pl == 2) ? py : px;
            const float gy = (pl == 0) ? py : pz;
            const float ix = (gx + 1.f) * 0.5f * (WW - 1);
            const float iy = (gy + 1.f) * 0.5f * (HH - 1);
            const float x0f = floorf(ix), y0f = floorf(iy);
            const float wx = ix - x0f, wy = iy - y0f;
            const int x0 = min(max((int)x0f,     0), WW - 1);
            const int x1 = min(max((int)x0f + 1, 0), WW - 1);
            const int y0 = min(max((int)y0f,     0), HH - 1);
            const int y1 = min(max((int)y0f + 1, 0), HH - 1);
            const float w00 = (1.f - wx) * (1.f - wy), w01 = wx * (1.f - wy);
            const float w10 = (1.f - wx) * wy,        w11 = wx * wy;
            const float* pb = triplane + (size_t)(b * 3 + pl) * CC * HH * WW;
            const int o00 = y0 * WW + x0, o01 = y0 * WW + x1;
            const int o10 = y1 * WW + x0, o11 = y1 * WW + x1;
            for (int c = cg * 32; c < cg * 32 + 32; ++c) {
                const float* cb = pb + (size_t)c * HH * WW;
                float v = cb[o00] * w00 + cb[o01] * w01 + cb[o10] * w10 + cb[o11] * w11;
                hbuf[pt * LDSK + pl * CC + c] = (_Float16)v;
            }
        }
    }
    __syncthreads();

    // ---- phase 2: fused MLP, wave-private 16-row M-tiles ----
    const int lane   = tid & 31;
    const int waveid = tid >> 5;           // 0..3
    const int lo16   = lane & 15;
    const int hi     = lane >> 4;
    const int mrow   = waveid * 16 + lo16;

    mlp_layer<K_IN>(w_in_h, b_in, hbuf, waveid, lo16, hi, mrow);
    for (int l = 0; l < NLAY - 1; ++l)
        mlp_layer<HID>(w_hid_h + (size_t)l * HID * HID, b_hid + l * HID,
                       hbuf, waveid, lo16, hi, mrow);

    // ---- phase 3: heads (rgb|den|emb packed as 16x256 matrix), one WMMA tile ----
    {
        half16 afr[8];
        const _Float16* arow = hbuf + mrow * LDSK;
        #pragma unroll
        for (int kf = 0; kf < 8; ++kf) {
            const int kb = kf * 32 + hi * 8;
            ((uint4*)&afr[kf])[0] = *(const uint4*)(arow + kb);
            ((uint4*)&afr[kf])[1] = *(const uint4*)(arow + kb + 16);
        }
        f32x8 acc = {0.f,0.f,0.f,0.f,0.f,0.f,0.f,0.f};
        const _Float16* wr = w_head_h + (size_t)lo16 * HID + hi * 8;
        #pragma unroll
        for (int kf = 0; kf < 8; ++kf) {
            half16 bfr;
            ((uint4*)&bfr)[0] = *(const uint4*)(wr + kf * 32);
            ((uint4*)&bfr)[1] = *(const uint4*)(wr + kf * 32 + 16);
            acc = __builtin_amdgcn_wmma_f32_16x16x32_f16(false, afr[kf], false, bfr,
                                                         (short)0, acc, false, false);
        }
        const int n = lo16;  // head output channel for this lane
        #pragma unroll
        for (int v = 0; v < 8; ++v) {
            const size_t p = (size_t)g0 + waveid * 16 + hi * 8 + v;  // global point
            const float r = acc[v];
            if (n < 3) {
                out[p * 3 + n] = 1.f / (1.f + __expf(-(r + b_rgb[n])));
            } else if (n == 3) {
                out[(size_t)BB * NN * 3 + p] = r + b_den[0];
            } else if (n < 7) {
                out[(size_t)BB * NN * 4 + p * 3 + (n - 4)] = r + b_emb[n - 4];
            }
        }
    }
}

extern "C" void kernel_launch(void* const* d_in, const int* in_sizes, int n_in,
                              void* d_out, int out_size, void* d_ws, size_t ws_size,
                              hipStream_t stream) {
    const float* triplane = (const float*)d_in[0];
    const float* points   = (const float*)d_in[1];
    const float* w_in     = (const float*)d_in[2];
    const float* b_in     = (const float*)d_in[3];
    const float* w_hid    = (const float*)d_in[4];
    const float* b_hid    = (const float*)d_in[5];
    const float* w_rgb    = (const float*)d_in[6];
    const float* b_rgb    = (const float*)d_in[7];
    const float* w_den    = (const float*)d_in[8];
    const float* b_den    = (const float*)d_in[9];
    const float* w_emb    = (const float*)d_in[10];
    const float* b_emb    = (const float*)d_in[11];

    _Float16* ws       = (_Float16*)d_ws;
    _Float16* w_in_h   = ws;
    _Float16* w_hid_h  = ws + W_IN_E;
    _Float16* w_head_h = ws + W_IN_E + W_HID_E;

    prep_weights<<<(W_TOTAL + 255) / 256, 256, 0, stream>>>(
        w_in, w_hid, w_rgb, w_den, w_emb, ws);

    triplane_mlp_kernel<<<(BB * NN) / TM, 128, 0, stream>>>(
        triplane, points, w_in_h, b_in, w_hid_h, b_hid, w_head_h,
        b_rgb, b_den, b_emb, (float*)d_out);
}
